// Transformer_25606595019086
// MI455X (gfx1250) — matrix operand order repaired and checked by reading, not compiled, not run on hardware
//
#include <hip/hip_runtime.h>
#include <hip/hip_bf16.h>
#include <math.h>

// ---------------- problem constants (from reference) ----------------
#define DD    512
#define NNTOK 4096
#define BBAT  2
#define HH    8
#define DHD   64
#define MLPD  2048
#define ROWS  (BBAT * NNTOK)   // 8192
#define TRIPLE (3 * HH * DHD)  // 1536

typedef _Float16 v16h __attribute__((ext_vector_type(16)));
typedef _Float16 v8h  __attribute__((ext_vector_type(8)));
typedef _Float16 v4h  __attribute__((ext_vector_type(4)));
typedef float    v8f  __attribute__((ext_vector_type(8)));
typedef float    v4f  __attribute__((ext_vector_type(4)));

union V16U { v16h v; v8h h[2]; };

__device__ __forceinline__ v8f wmma_f32_f16(v16h a, v16h b, v8f c) {
  return __builtin_amdgcn_wmma_f32_16x16x32_f16(false, a, false, b, (short)0, c,
                                                false, false);
}

// A-operand fragment from LDS row (16x32 A tile; this code's contiguous k order, same as B):
// lane L: M = L%16 ; halves 0..15 = K[lh*16 .. +16] contiguous
__device__ __forceinline__ v16h lds_afrag(const _Float16* rowp, int lh) {
  V16U u;
  u.h[0] = *(const v8h*)(rowp + lh * 16);
  u.h[1] = *(const v8h*)(rowp + lh * 16 + 8);
  return u.v;
}

// B-operand fragment from LDS (Bt stored N-major): lane n: col = n%16,
// K = 16*(n/16) .. +16 contiguous.
__device__ __forceinline__ v16h lds_bfrag(const _Float16* p) {
  V16U u;
  u.h[0] = *(const v8h*)(p);
  u.h[1] = *(const v8h*)(p + 8);
  return u.v;
}

// =====================================================================
// WMMA GEMM: C[M,N] = A[M,K](f16) @ B[K,N](f16), templated epilogue.
// FLAGS: 1=+bias  2=gelu(exact)  4=+resid(f32);  OUTH: write f16 else f32.
// block 256 (8 waves, 2x4 grid), tile 128x128, wave tile 64x32, K-slab 32.
// Global->reg prefetch of the next slab overlaps VMEM with WMMA; L2
// prefetch (global_prefetch_b8) runs two slabs ahead.  All fragment loads
// are hoisted ahead of the WMMA burst so dscnt waits stagger.
// =====================================================================
#define LDA 40   // 32 + 8 halfs pad
template <int FLAGS, bool OUTH>
__launch_bounds__(256)
__global__ void gemm_f16_wmma(const _Float16* __restrict__ A,
                              const _Float16* __restrict__ Bw,
                              const float* __restrict__ bias,
                              const float* __restrict__ resid,
                              float* __restrict__ outF,
                              _Float16* __restrict__ outH,
                              int M, int N, int K) {
  __shared__ _Float16 As[128 * LDA];
  __shared__ _Float16 Bts[128 * LDA];

  const int m0 = blockIdx.y * 128;
  const int n0 = blockIdx.x * 128;
  const int tid  = threadIdx.x;
  const int wid  = tid >> 5;
  const int lane = tid & 31;
  const int lr = lane & 15, lh = lane >> 4;
  const int wmb = (wid >> 2) * 64;   // wave M offset (0/64)
  const int wnb = (wid & 3) * 32;    // wave N offset (0..96)
  const int kg = tid >> 5;           // B micro-tile: 4 k-rows
  const int ng = tid & 31;           //               4 n-cols

  v8f acc[4][2] = {};
  v8h apre[2];
  v4h bpre[4];

  auto gload = [&](int k0) {
#pragma unroll
    for (int i = 0; i < 2; i++) {   // A: 128x32 tile, 512 16B-chunks
      int idx = tid + i * 256;
      int row = idx >> 2, g = idx & 3;
      apre[i] = *(const v8h*)(A + (size_t)(m0 + row) * K + k0 + g * 8);
    }
#pragma unroll
    for (int j = 0; j < 4; j++)     // B: 4(k) x 4(n) micro-tile per thread
      bpre[j] = *(const v4h*)(Bw + (size_t)(k0 + kg * 4 + j) * N + n0 + ng * 4);
    if (k0 + 32 < K) {              // L2 prefetch two slabs ahead
      __builtin_prefetch(A + (size_t)(m0 + (tid >> 2)) * K + k0 + 32, 0, 1);
      __builtin_prefetch(Bw + (size_t)(k0 + 32 + kg) * N + n0 + ng * 4, 0, 1);
    }
  };

  gload(0);
  for (int k0 = 0; k0 < K; k0 += 32) {
    // commit staged registers to LDS (B transposed in-register -> Bts[n][k])
#pragma unroll
    for (int i = 0; i < 2; i++) {
      int idx = tid + i * 256;
      int row = idx >> 2, g = idx & 3;
      *(v8h*)&As[row * LDA + g * 8] = apre[i];
    }
#pragma unroll
    for (int d = 0; d < 4; d++) {
      v4h t = {bpre[0][d], bpre[1][d], bpre[2][d], bpre[3][d]};
      *(v4h*)&Bts[(ng * 4 + d) * LDA + kg * 4] = t;
    }
    __syncthreads();

    if (k0 + 32 < K) gload(k0 + 32);  // overlaps with WMMAs below

    v16h bf0 = lds_bfrag(&Bts[(wnb + lr) * LDA + 16 * lh]);
    v16h bf1 = lds_bfrag(&Bts[(wnb + 16 + lr) * LDA + 16 * lh]);
    v16h af[4];
#pragma unroll
    for (int mi = 0; mi < 4; mi++)
      af[mi] = lds_afrag(&As[(wmb + mi * 16 + lr) * LDA], lh);
#pragma unroll
    for (int mi = 0; mi < 4; mi++) {
      acc[mi][0] = wmma_f32_f16(af[mi], bf0, acc[mi][0]);
      acc[mi][1] = wmma_f32_f16(af[mi], bf1, acc[mi][1]);
    }
    __syncthreads();
  }

  // epilogue: C layout lane n: VGPR r -> (M = r + 8*(n/16), N = n%16)
#pragma unroll
  for (int mi = 0; mi < 4; mi++)
#pragma unroll
    for (int ni = 0; ni < 2; ni++)
#pragma unroll
      for (int r = 0; r < 8; r++) {
        int row = m0 + wmb + mi * 16 + r + 8 * lh;
        int col = n0 + wnb + ni * 16 + lr;
        float v = acc[mi][ni][r];
        if (FLAGS & 1) v += bias[col];
        if (FLAGS & 2) v = 0.5f * v * (1.0f + erff(v * 0.70710678118654752f));
        size_t idx = (size_t)row * N + col;
        if (FLAGS & 4) v += resid[idx];
        if (OUTH) outH[idx] = (_Float16)v;
        else      outF[idx] = v;
      }
}

// =====================================================================
// Flash attention: qkv f16 [B,N,1536] -> out f16 [B,N,512] (head-major).
// grid(N/64, H, B), block 128 (4 waves, 16 q-rows each). KV chunk = 32.
// =====================================================================
#define LDK 72
#define LDV 40
#define LDP 40
__launch_bounds__(128)
__global__ void flash_attn(const _Float16* __restrict__ qkv,
                           _Float16* __restrict__ out) {
  __shared__ _Float16 Ks[32 * LDK];       // K rows as-is  (Bt for Q@K^T)
  __shared__ _Float16 Vt[64 * LDV];       // V transposed  (Bt for P@V)
  __shared__ _Float16 Ps[4 * 16 * LDP];   // per-wave P tile (A operand)

  const int bb = blockIdx.z, hh = blockIdx.y;
  const int tid = threadIdx.x, wv = tid >> 5, lane = tid & 31;
  const int lr = lane & 15, lh = lane >> 4;
  const size_t bN = (size_t)bb * NNTOK;
  const int q0 = blockIdx.x * 64 + wv * 16;
  const float scale = 0.125f; // DH^-0.5
  const int kvg = tid >> 4;   // V micro-tile: 4 kv-rows
  const int dhg = tid & 15;   //               4 dh-cols

  // Q fragments kept in registers for whole kernel (DH=64 -> 2 K-steps)
  const _Float16* qrow = qkv + (bN + q0 + lr) * TRIPLE + hh * DHD;
  V16U q0u, q1u;
  q0u.h[0] = *(const v8h*)(qrow + lh * 16);
  q0u.h[1] = *(const v8h*)(qrow + lh * 16 + 8);
  q1u.h[0] = *(const v8h*)(qrow + 32 + lh * 16);
  q1u.h[1] = *(const v8h*)(qrow + 40 + lh * 16);
  const v16h qf0 = q0u.v, qf1 = q1u.v;

  float mrow[8], lrow[8];
#pragma unroll
  for (int r = 0; r < 8; r++) { mrow[r] = -1e30f; lrow[r] = 0.0f; }
  v8f o[4] = {};

  for (int kc = 0; kc < NNTOK; kc += 32) {
    __syncthreads();
#pragma unroll
    for (int i = 0; i < 2; i++) { // stage 32 K rows (row-major, 16B chunks)
      int idx = tid + i * 128;    // 0..255
      int r = idx >> 3, g = idx & 7;
      const _Float16* src = qkv + (bN + kc + r) * TRIPLE + hh * DHD;
      *(v8h*)&Ks[r * LDK + g * 8] = *(const v8h*)(src + 512 + g * 8);
      if (kc + 32 < NNTOK)
        __builtin_prefetch(src + 32 * TRIPLE + 512 + g * 8, 0, 1);
    }
    { // stage 32 V rows transposed: 4(kv) x 4(dh) micro-tile per thread
      v4h vv[4];
#pragma unroll
      for (int j = 0; j < 4; j++) {
        const _Float16* src = qkv + (bN + kc + kvg * 4 + j) * TRIPLE + 1024 +
                              hh * DHD + dhg * 4;
        vv[j] = *(const v4h*)src;
        if (kc + 32 < NNTOK) __builtin_prefetch(src + 32 * TRIPLE, 0, 1);
      }
#pragma unroll
      for (int d = 0; d < 4; d++) {
        v4h t = {vv[0][d], vv[1][d], vv[2][d], vv[3][d]};
        *(v4h*)&Vt[(dhg * 4 + d) * LDV + kvg * 4] = t;
      }
    }
    __syncthreads();

    // S = Q @ K^T  (two 16-col tiles, two chained K-steps over DH);
    // hoist all four K fragments so dscnt waits stagger.
    v16h kf00 = lds_bfrag(&Ks[lr * LDK + 16 * lh]);
    v16h kf01 = lds_bfrag(&Ks[lr * LDK + 32 + 16 * lh]);
    v16h kf10 = lds_bfrag(&Ks[(16 + lr) * LDK + 16 * lh]);
    v16h kf11 = lds_bfrag(&Ks[(16 + lr) * LDK + 32 + 16 * lh]);
    v8f z = {};
    v8f s0 = wmma_f32_f16(qf0, kf00, z);
    v8f s1 = wmma_f32_f16(qf0, kf10, z);
    s0 = wmma_f32_f16(qf1, kf01, s0);
    s1 = wmma_f32_f16(qf1, kf11, s1);

    // online softmax per q-row (rows r+8*lh live on the 16 lanes sharing lh)
#pragma unroll
    for (int r = 0; r < 8; r++) {
      float a = s0[r] * scale, c = s1[r] * scale;
      float mx = fmaxf(a, c);
#pragma unroll
      for (int d = 1; d < 16; d <<= 1) mx = fmaxf(mx, __shfl_xor(mx, d, 32));
      float mn = fmaxf(mrow[r], mx);
      float al = __expf(mrow[r] - mn);
      mrow[r] = mn;
      float p0 = __expf(a - mn), p1 = __expf(c - mn);
      float rs = p0 + p1;
#pragma unroll
      for (int d = 1; d < 16; d <<= 1) rs += __shfl_xor(rs, d, 32);
      lrow[r] = lrow[r] * al + rs;
      Ps[(wv * 16 + r + 8 * lh) * LDP + lr] = (_Float16)p0;
      Ps[(wv * 16 + r + 8 * lh) * LDP + 16 + lr] = (_Float16)p1;
#pragma unroll
      for (int j = 0; j < 4; j++) o[j][r] *= al;
    }
    __syncthreads();

    // O += P @ V   (P: 16x32 A-operand from LDS; V: 32x64 via Vt)
    v16h pf = lds_afrag(&Ps[(wv * 16 + lr) * LDP], lh);
    v16h vf[4];
#pragma unroll
    for (int j = 0; j < 4; j++)
      vf[j] = lds_bfrag(&Vt[(16 * j + lr) * LDV + 16 * lh]);
#pragma unroll
    for (int j = 0; j < 4; j++) o[j] = wmma_f32_f16(pf, vf[j], o[j]);
  }

#pragma unroll
  for (int j = 0; j < 4; j++)
#pragma unroll
    for (int r = 0; r < 8; r++) {
      float v = o[j][r] / lrow[r];
      out[(bN + q0 + r + 8 * lh) * 512 + hh * DHD + 16 * j + lr] = (_Float16)v;
    }
}

// =====================================================================
// LayerNorm over D=512: one wave per row, 8 rows per 256-thread block.
// =====================================================================
__launch_bounds__(256)
__global__ void layernorm_k(const float* __restrict__ x,
                            const float* __restrict__ g,
                            const float* __restrict__ bta,
                            _Float16* __restrict__ outH,
                            float* __restrict__ outF) {
  const int row = blockIdx.x * 8 + (threadIdx.x >> 5);
  const int lane = threadIdx.x & 31;
  const float* xr = x + (size_t)row * DD;
  float v[16];
  float s = 0.f, ss = 0.f;
#pragma unroll
  for (int i = 0; i < 16; i++) {
    v[i] = xr[lane + i * 32];
    s += v[i];
    ss += v[i] * v[i];
  }
#pragma unroll
  for (int d = 1; d < 32; d <<= 1) {
    s += __shfl_xor(s, d, 32);
    ss += __shfl_xor(ss, d, 32);
  }
  const float mean = s * (1.0f / DD);
  const float var = ss * (1.0f / DD) - mean * mean;
  const float rstd = rsqrtf(var + 1e-5f);
#pragma unroll
  for (int i = 0; i < 16; i++) {
    int c = lane + i * 32;
    float o = (v[i] - mean) * rstd * g[c] + bta[c];
    if (outH) outH[(size_t)row * DD + c] = (_Float16)o;
    if (outF) outF[(size_t)row * DD + c] = o;
  }
}

// fp32 -> f16 elementwise (weights, skip snapshots); n4 = n/4
__global__ void cvt_f32_f16(const float* __restrict__ src,
                            _Float16* __restrict__ dst, int n4) {
  int i = blockIdx.x * 256 + threadIdx.x;
  if (i < n4) {
    v4f f = *(const v4f*)(src + (size_t)i * 4);
    v4h hv;
#pragma unroll
    for (int j = 0; j < 4; j++) hv[j] = (_Float16)f[j];
    *(v4h*)(dst + (size_t)i * 4) = hv;
  }
}

// cc[row, 0:512] = skip(f16) ; cc[row, 512:1024] = f16(x)
__global__ void pack_concat(const _Float16* __restrict__ skip,
                            const float* __restrict__ x,
                            _Float16* __restrict__ cc) {
  int t = blockIdx.x * 256 + threadIdx.x; // ROWS*128 threads, 4 cols each
  int row = t >> 7, cq = (t & 127) * 4;
  *(v4h*)(cc + (size_t)row * 1024 + cq) =
      *(const v4h*)(skip + (size_t)row * 512 + cq);
  v4f f = *(const v4f*)(x + (size_t)row * 512 + cq);
  v4h hv;
#pragma unroll
  for (int j = 0; j < 4; j++) hv[j] = (_Float16)f[j];
  *(v4h*)(cc + (size_t)row * 1024 + 512 + cq) = hv;
}

// =====================================================================
// Host orchestration
// =====================================================================
extern "C" void kernel_launch(void* const* d_in, const int* in_sizes, int n_in,
                              void* d_out, int out_size, void* d_ws,
                              size_t ws_size, hipStream_t stream) {
  const float* in_x = (const float*)d_in[0];
  const float* cw   = (const float*)d_in[1];
  const float* cb   = (const float*)d_in[2];
  const float* ag   = (const float*)d_in[3];
  const float* abp  = (const float*)d_in[4];
  const float* qkvw = (const float*)d_in[5];
  const float* oww  = (const float*)d_in[6];
  const float* fg   = (const float*)d_in[7];
  const float* fbp  = (const float*)d_in[8];
  const float* w1   = (const float*)d_in[9];
  const float* b1   = (const float*)d_in[10];
  const float* w2   = (const float*)d_in[11];
  const float* b2   = (const float*)d_in[12];
  const float* fing = (const float*)d_in[13];
  const float* finb = (const float*)d_in[14];

  char* p = (char*)d_ws;
  auto carve = [&](size_t bytes) -> void* {
    void* r = (void*)p;
    p += (bytes + 255) & ~(size_t)255;
    return r;
  };
  _Float16* hcw  = (_Float16*)carve((size_t)8 * 1024 * 512 * 2);
  _Float16* hqkv = (_Float16*)carve((size_t)8 * 512 * TRIPLE * 2);
  _Float16* how  = (_Float16*)carve((size_t)8 * 512 * 512 * 2);
  _Float16* hw1  = (_Float16*)carve((size_t)8 * 512 * MLPD * 2);
  _Float16* hw2  = (_Float16*)carve((size_t)8 * MLPD * 512 * 2);
  float*    xbuf = (float*)carve((size_t)ROWS * 512 * 4);
  _Float16* xn   = (_Float16*)carve((size_t)ROWS * 512 * 2);
  _Float16* ccb  = (_Float16*)carve((size_t)ROWS * 1024 * 2);
  _Float16* qkvb = (_Float16*)carve((size_t)ROWS * TRIPLE * 2);
  _Float16* aout = (_Float16*)carve((size_t)ROWS * 512 * 2);
  _Float16* h1   = (_Float16*)carve((size_t)ROWS * MLPD * 2);
  _Float16* skips[4];
  for (int i = 0; i < 4; i++)
    skips[i] = (_Float16*)carve((size_t)ROWS * 512 * 2);

  auto cvt = [&](const float* s, _Float16* d, size_t n) {
    int n4 = (int)(n / 4);
    cvt_f32_f16<<<(n4 + 255) / 256, 256, 0, stream>>>(s, d, n4);
  };
  cvt(cw, hcw, (size_t)8 * 1024 * 512);
  cvt(qkvw, hqkv, (size_t)8 * 512 * TRIPLE);
  cvt(oww, how, (size_t)8 * 512 * 512);
  cvt(w1, hw1, (size_t)8 * 512 * MLPD);
  cvt(w2, hw2, (size_t)8 * MLPD * 512);

  hipMemcpyAsync(xbuf, in_x, (size_t)ROWS * 512 * 4, hipMemcpyDeviceToDevice,
                 stream);

  const dim3 g512(512 / 128, ROWS / 128);
  const dim3 g1536(TRIPLE / 128, ROWS / 128);
  const dim3 g2048(MLPD / 128, ROWS / 128);

  int sp = 0;
  for (int ind = 0; ind < 8; ind++) {
    const int layer = ind + 1;
    if (layer <= 4) { // push skip snapshot (as f16, used only by combine GEMM)
      cvt(xbuf, skips[sp], (size_t)ROWS * 512);
      sp++;
    }
    if (layer >= 5) { // pop + skip-combine linear
      sp--;
      pack_concat<<<ROWS * 128 / 256, 256, 0, stream>>>(skips[sp], xbuf, ccb);
      gemm_f16_wmma<1, false><<<g512, 256, 0, stream>>>(
          ccb, hcw + (size_t)ind * 1024 * 512, cb + (size_t)ind * 512, nullptr,
          xbuf, nullptr, ROWS, 512, 1024);
    }
    // ---- attention ----
    layernorm_k<<<ROWS / 8, 256, 0, stream>>>(xbuf, ag + (size_t)ind * 512,
                                              abp + (size_t)ind * 512, xn,
                                              nullptr);
    gemm_f16_wmma<0, true><<<g1536, 256, 0, stream>>>(
        xn, hqkv + (size_t)ind * 512 * TRIPLE, nullptr, nullptr, nullptr, qkvb,
        ROWS, TRIPLE, 512);
    flash_attn<<<dim3(NNTOK / 64, HH, BBAT), 128, 0, stream>>>(qkvb, aout);
    gemm_f16_wmma<4, false><<<g512, 256, 0, stream>>>(
        aout, how + (size_t)ind * 512 * 512, nullptr, xbuf, xbuf, nullptr,
        ROWS, 512, 512);
    // ---- feed-forward ----
    layernorm_k<<<ROWS / 8, 256, 0, stream>>>(xbuf, fg + (size_t)ind * 512,
                                              fbp + (size_t)ind * 512, xn,
                                              nullptr);
    gemm_f16_wmma<3, true><<<g2048, 256, 0, stream>>>(
        xn, hw1 + (size_t)ind * 512 * MLPD, b1 + (size_t)ind * MLPD, nullptr,
        nullptr, h1, ROWS, MLPD, 512);
    gemm_f16_wmma<5, false><<<g512, 256, 0, stream>>>(
        h1, hw2 + (size_t)ind * MLPD * 512, b2 + (size_t)ind * 512, xbuf, xbuf,
        nullptr, ROWS, 512, MLPD);
  }
  // final norm -> d_out (fp32)
  layernorm_k<<<ROWS / 8, 256, 0, stream>>>(xbuf, fing, finb, nullptr,
                                            (float*)d_out);
  (void)in_sizes; (void)n_in; (void)out_size; (void)ws_size;
}
